// InContextAssoMemBlock_45578192945718
// MI455X (gfx1250) — matrix operand order repaired and checked by reading, not compiled, not run on hardware
//
#include <hip/hip_runtime.h>

#define B_  2
#define T_  2048
#define NH_ 16
#define HS_ 64
#define C_  1024
#define EXP_SCALE_ 10.0f
#define KB_UP_ 5.0f

typedef __attribute__((ext_vector_type(16))) _Float16 v16h;
typedef __attribute__((ext_vector_type(8)))  float    v8f;

#if __has_builtin(__builtin_amdgcn_global_load_async_to_lds_b128) && \
    __has_builtin(__builtin_amdgcn_s_wait_asynccnt)
#define HAVE_ASYNC_LDS 1
#else
#define HAVE_ASYNC_LDS 0
#endif

#if HAVE_ASYNC_LDS
typedef int v4i_async __attribute__((__vector_size__(4 * sizeof(int))));
typedef __attribute__((address_space(1))) v4i_async* async_gptr;
typedef __attribute__((address_space(3))) v4i_async* async_lptr;
#endif

// 16-byte global -> LDS copy. Async DMA (ASYNCcnt) when available, else via VGPRs.
static __device__ __forceinline__ void copy16(const _Float16* g, _Float16* l) {
#if HAVE_ASYNC_LDS
  __builtin_amdgcn_global_load_async_to_lds_b128(
      (async_gptr)(void*)g, (async_lptr)(void*)l, 0, 0);
#else
  *(uint4*)l = *(const uint4*)g;
#endif
}

// Wait for previous tile's async loads. Each thread issues exactly 4 async-load
// instructions per tile; completions are in-order, so with a next tile in
// flight (4 outstanding) a threshold of 4 retires the previous tile.
static __device__ __forceinline__ void async_wait(bool next_in_flight) {
#if HAVE_ASYNC_LDS
  if (next_in_flight) __builtin_amdgcn_s_wait_asynccnt(4);
  else                __builtin_amdgcn_s_wait_asynccnt(0);
#else
  (void)next_in_flight;
#endif
}

static __device__ __forceinline__ v8f wmma_f16(v16h a, v16h b, v8f c) {
  // D = A(16x32 f16) x B(32x16 f16) + C(16x16 f32)
  return __builtin_amdgcn_wmma_f32_16x16x32_f16(false, a, false, b, (short)0, c, false, false);
}

// ---- Fragment loaders (wave32); A and B share contiguous order k = 16h + e, not the ISA table order ----

// A fragment: 16(M) x 32(K) f16; LDS tile row-major, row stride ldm halves.
// lane l: m = l&15, h = l>>4 ; element e: K = 16h + e
static __device__ __forceinline__ v16h load_frag_a(const _Float16* p, int ldm) {
  int l = threadIdx.x & 31;
  int m = l & 15, h = l >> 4;
  const _Float16* row = p + m * ldm + 16 * h;
  v16h a;
  #pragma unroll
  for (int e = 0; e < 8; ++e)  a[e] = row[e];
  #pragma unroll
  for (int e = 8; e < 16; ++e) a[e] = row[e];
  return a;
}

// B fragment: 32(K) x 16(N) f16; LDS tile stored K-major with stride ldk halves.
// lane l: n = l&15, h = l>>4 ; element e: K = 16h + e
static __device__ __forceinline__ v16h load_frag_b(const _Float16* p, int ldk) {
  int l = threadIdx.x & 31;
  int n = l & 15, h = l >> 4;
  const _Float16* base = p + (16 * h) * ldk + n;
  v16h b;
  #pragma unroll
  for (int e = 0; e < 16; ++e) b[e] = base[e * ldk];
  return b;
}

// B fragment where the LDS tile is stored N-major: value(k,n) = p[n*ldn + k]
// (used for K^T in the score GEMM: keys are rows of sK, contraction over HS)
static __device__ __forceinline__ v16h load_frag_bT(const _Float16* p, int ldn) {
  int l = threadIdx.x & 31;
  int n = l & 15, h = l >> 4;
  const _Float16* base = p + n * ldn + 16 * h;
  v16h b;
  #pragma unroll
  for (int e = 0; e < 16; ++e) b[e] = base[e];
  return b;
}

// -------------------------------------------------------------------------

__global__ void cast_f32_to_f16(const float* __restrict__ in, _Float16* __restrict__ out, int n) {
  int i = blockIdx.x * blockDim.x + threadIdx.x;
  if (i < n) out[i] = (_Float16)in[i];
}

// GEMM: Out(MxN,f32) = A(MxK,f16) x Bw(KxN,f16). Block tile 128x128, 8 waves
// arranged 2(M) x 4(N); each wave owns a 64x32 sub-tile = 4x2 WMMA tiles.
// Double-buffered LDS tiles; async DMA staging of tile s+1 overlaps WMMAs of s.
__global__ __launch_bounds__(256) void gemm_f16_f32_wmma(
    const _Float16* __restrict__ A, const _Float16* __restrict__ Bw,
    float* __restrict__ Out, int M, int N, int K)
{
  __shared__ __align__(16) _Float16 sA[2][128 * 32];
  __shared__ __align__(16) _Float16 sB[2][32 * 128];
  const int tid  = threadIdx.x;
  const int wave = tid >> 5;
  const int wm = wave >> 2, wn = wave & 3;
  const int bm = blockIdx.y * 128, bn = blockIdx.x * 128;

  auto stageA = [&](int buf, int k0) {  // 128x32 halves, 2 async b128 per thread
    for (int i = tid; i < 512; i += 256) {
      int r = i >> 2, c = (i & 3) * 8;
      copy16(A + (size_t)(bm + r) * K + k0 + c, &sA[buf][r * 32 + c]);
    }
  };
  auto stageB = [&](int buf, int k0) {  // 32x128 halves, 2 async b128 per thread
    for (int i = tid; i < 512; i += 256) {
      int r = i >> 4, c = (i & 15) * 8;
      copy16(Bw + (size_t)(k0 + r) * N + bn + c, &sB[buf][r * 128 + c]);
    }
  };

  v8f zero = {};
  v8f acc[4][2];
  #pragma unroll
  for (int i = 0; i < 4; ++i)
    #pragma unroll
    for (int j = 0; j < 2; ++j) acc[i][j] = zero;

  const int nsteps = K / 32;
  stageA(0, 0);
  stageB(0, 0);
  for (int s = 0; s < nsteps; ++s) {
    const int cur = s & 1;
    const bool pre = (s + 1 < nsteps);
    if (pre) { stageA(cur ^ 1, (s + 1) * 32); stageB(cur ^ 1, (s + 1) * 32); }
#if !HAVE_ASYNC_LDS
    if (pre)  // hint the next tile toward L2 (global_prefetch_b8)
      __builtin_prefetch(A + (size_t)(bm + (tid >> 1)) * K + (s + 1) * 32, 0, 1);
#endif
    async_wait(pre);
    __syncthreads();

    v16h bf[2];
    #pragma unroll
    for (int j = 0; j < 2; ++j)
      bf[j] = load_frag_b(&sB[cur][wn * 32 + j * 16], 128);
    #pragma unroll
    for (int i = 0; i < 4; ++i) {
      v16h af = load_frag_a(&sA[cur][(wm * 64 + i * 16) * 32], 32);
      #pragma unroll
      for (int j = 0; j < 2; ++j)
        acc[i][j] = wmma_f16(af, bf[j], acc[i][j]);
    }
    __syncthreads();
  }

  const int l = tid & 31;
  const int col = l & 15, rh = (l >> 4) * 8;
  #pragma unroll
  for (int i = 0; i < 4; ++i)
    #pragma unroll
    for (int j = 0; j < 2; ++j) {
      int gm = bm + wm * 64 + i * 16 + rh;
      int gn = bn + wn * 32 + j * 16 + col;
      #pragma unroll
      for (int r = 0; r < 8; ++r)
        Out[(size_t)(gm + r) * N + gn] = acc[i][j][r];
    }
}

// EMA scan over time + L2-normalize + beta scale -> k (f16, [B,NH,T,HS])
__global__ __launch_bounds__(32) void ema_norm_k(
    const float* __restrict__ xl, const float* __restrict__ la_coef,
    const float* __restrict__ kernel_beta, _Float16* __restrict__ Kh)
{
  const int bh = blockIdx.x;
  const int b = bh / NH_, h = bh % NH_;
  const int lane = threadIdx.x;
  const float alpha = la_coef[h];
  const float kb = __expf(fminf(kernel_beta[h] * EXP_SCALE_, KB_UP_));
  const float* src = xl + (size_t)b * T_ * C_ + h * HS_;
  _Float16* dst = Kh + (size_t)bh * T_ * HS_;
  float c0 = 0.f, c1 = 0.f;
  for (int t = 0; t < T_; ++t) {
    float x0 = src[(size_t)t * C_ + lane];
    float x1 = src[(size_t)t * C_ + lane + 32];
    c0 = alpha * c0 + (1.f - alpha) * x0;
    c1 = alpha * c1 + (1.f - alpha) * x1;
    float s = c0 * c0 + c1 * c1;
    #pragma unroll
    for (int d = 1; d < 32; d <<= 1) s += __shfl_xor(s, d, 32);
    float rn = kb * rsqrtf(s);
    dst[(size_t)t * HS_ + lane]      = (_Float16)(c0 * rn);
    dst[(size_t)t * HS_ + lane + 32] = (_Float16)(c1 * rn);
  }
}

// v = normalize(xv[t+1]*(1-c) + xv[t]*c) * exp(value_beta*10) -> f16 [B,NH,T,HS]
__global__ __launch_bounds__(256) void vmix_norm_k(
    const float* __restrict__ xv, const float* __restrict__ v_coef,
    const float* __restrict__ value_beta, _Float16* __restrict__ Vh)
{
  const int gw = (int)((blockIdx.x * blockDim.x + threadIdx.x) >> 5);
  const int lane = threadIdx.x & 31;
  if (gw >= B_ * NH_ * T_) return;
  const int t = gw % T_;
  const int bh = gw / T_;
  const int b = bh / NH_, h = bh % NH_;
  const float c = v_coef[h];
  const float vb = __expf(value_beta[h] * EXP_SCALE_);
  const float* base = xv + (size_t)b * T_ * C_ + h * HS_;
  float x0 = base[(size_t)t * C_ + lane];
  float x1 = base[(size_t)t * C_ + lane + 32];
  float xs0 = 0.f, xs1 = 0.f;
  if (t + 1 < T_) {
    xs0 = base[(size_t)(t + 1) * C_ + lane];
    xs1 = base[(size_t)(t + 1) * C_ + lane + 32];
  }
  float v0 = xs0 * (1.f - c) + x0 * c;
  float v1 = xs1 * (1.f - c) + x1 * c;
  float s = v0 * v0 + v1 * v1;
  #pragma unroll
  for (int d = 1; d < 32; d <<= 1) s += __shfl_xor(s, d, 32);
  float rn = vb * rsqrtf(s);
  Vh[((size_t)bh * T_ + t) * HS_ + lane]      = (_Float16)(v0 * rn);
  Vh[((size_t)bh * T_ + t) * HS_ + lane + 32] = (_Float16)(v1 * rn);
}

// Causal flash attention with WMMA. q_t = k_t, keys j < t, out row t=0 is zero.
// Block: 64 q rows (4 waves x 16). Double-buffered K/V tiles of 32 keys staged
// into LDS with async DMA; staging of tile s+1 overlaps WMMAs of tile s.
__global__ __launch_bounds__(128) void flash_attn_wmma(
    const _Float16* __restrict__ Kh, const _Float16* __restrict__ Vh,
    _Float16* __restrict__ Yh)
{
  __shared__ __align__(16) _Float16 sK[2][32 * 64];
  __shared__ __align__(16) _Float16 sV[2][32 * 64];
  __shared__ __align__(16) _Float16 sP[4][16 * 32];

  const int bh = blockIdx.y;
  const int b = bh / NH_, h = bh % NH_;
  const int qb = blockIdx.x * 64;
  const int wave = threadIdx.x >> 5;
  const int lane = threadIdx.x & 31;
  const int qbase = qb + wave * 16;

  const _Float16* Kp = Kh + (size_t)bh * T_ * HS_;
  const _Float16* Vp = Vh + (size_t)bh * T_ * HS_;

  auto stageKV = [&](int buf, int j0) {  // 2x 32x64 halves, 4 async b128/thread
    for (int i = threadIdx.x; i < 256; i += 128) {
      int r = i >> 3, c = (i & 7) * 8;
      copy16(Kp + (size_t)(j0 + r) * HS_ + c, &sK[buf][r * 64 + c]);
      copy16(Vp + (size_t)(j0 + r) * HS_ + c, &sV[buf][r * 64 + c]);
    }
  };

  // Q fragments for rows qbase..qbase+15, HS split into two K=32 chunks
  v16h qa0, qa1;
  {
    int m = lane & 15, hh = lane >> 4;
    const _Float16* qrow = Kp + (size_t)(qbase + m) * HS_ + 16 * hh;
    #pragma unroll
    for (int e = 0; e < 8; ++e)  { qa0[e] = qrow[e];     qa1[e] = qrow[32 + e]; }
    #pragma unroll
    for (int e = 8; e < 16; ++e) { qa0[e] = qrow[e]; qa1[e] = qrow[32 + e]; }
  }

  v8f zero = {};
  v8f o0 = zero, o1 = zero, o2 = zero, o3 = zero;
  float m8[8], l8[8];
  #pragma unroll
  for (int r = 0; r < 8; ++r) { m8[r] = -1e30f; l8[r] = 0.f; }

  const int colk = lane & 15;
  const int rowh = (lane >> 4) * 8;

  const int nsteps = (qb + 64) / 32;
  stageKV(0, 0);
  for (int s = 0; s < nsteps; ++s) {
    const int cur = s & 1;
    const bool pre = (s + 1 < nsteps);
    const int j0 = s * 32;
    if (pre) stageKV(cur ^ 1, (s + 1) * 32);
    async_wait(pre);
    __syncthreads();

    const _Float16* kT = sK[cur];
    const _Float16* vT = sV[cur];

    // scores: two 16x16 tiles (keys j0..j0+15 and j0+16..j0+31), contract HS=64
    v8f s0 = zero, s1 = zero;
    s0 = wmma_f16(qa0, load_frag_bT(kT, 64), s0);
    s0 = wmma_f16(qa1, load_frag_bT(kT + 32, 64), s0);
    s1 = wmma_f16(qa0, load_frag_bT(kT + 16 * 64, 64), s1);
    s1 = wmma_f16(qa1, load_frag_bT(kT + 16 * 64 + 32, 64), s1);

    if (j0 + 32 > qbase) {  // diagonal-overlapping step: causal mask (key < t)
      #pragma unroll
      for (int r = 0; r < 8; ++r) {
        int tq = qbase + rowh + r;
        if (j0 + colk      >= tq) s0[r] = -1e30f;
        if (j0 + 16 + colk >= tq) s1[r] = -1e30f;
      }
    }

    // online softmax; row stats reduced across the 16 lanes holding each row
    #pragma unroll
    for (int r = 0; r < 8; ++r) {
      float mx = fmaxf(s0[r], s1[r]);
      #pragma unroll
      for (int d = 1; d < 16; d <<= 1) mx = fmaxf(mx, __shfl_xor(mx, d, 32));
      float mnew = fmaxf(m8[r], mx);
      float scale = __expf(m8[r] - mnew);
      float p0 = __expf(s0[r] - mnew);
      float p1 = __expf(s1[r] - mnew);
      float ps = p0 + p1;
      #pragma unroll
      for (int d = 1; d < 16; d <<= 1) ps += __shfl_xor(ps, d, 32);
      l8[r] = l8[r] * scale + ps;
      m8[r] = mnew;
      o0[r] *= scale; o1[r] *= scale; o2[r] *= scale; o3[r] *= scale;
      // repack P (D layout) -> LDS so it can be reloaded in A-fragment layout
      sP[wave][(rowh + r) * 32 + colk]      = (_Float16)p0;
      sP[wave][(rowh + r) * 32 + 16 + colk] = (_Float16)p1;
    }

    // O += P(16x32) x V(32x64); vT is already K-major for the B fragments
    v16h pa = load_frag_a(&sP[wave][0], 32);
    o0 = wmma_f16(pa, load_frag_b(vT,      64), o0);
    o1 = wmma_f16(pa, load_frag_b(vT + 16, 64), o1);
    o2 = wmma_f16(pa, load_frag_b(vT + 32, 64), o2);
    o3 = wmma_f16(pa, load_frag_b(vT + 48, 64), o3);
    __syncthreads();
  }

  // write y (B,T,C) layout as f16 for the projection GEMM; t=0 row forced zero
  _Float16* yrow = Yh + ((size_t)b * T_) * C_ + h * HS_;
  #pragma unroll
  for (int r = 0; r < 8; ++r) {
    int tq = qbase + rowh + r;
    float inv = (tq == 0) ? 0.f : 1.f / l8[r];
    yrow[(size_t)tq * C_ +      colk] = (_Float16)(o0[r] * inv);
    yrow[(size_t)tq * C_ + 16 + colk] = (_Float16)(o1[r] * inv);
    yrow[(size_t)tq * C_ + 32 + colk] = (_Float16)(o2[r] * inv);
    yrow[(size_t)tq * C_ + 48 + colk] = (_Float16)(o3[r] * inv);
  }
}

extern "C" void kernel_launch(void* const* d_in, const int* in_sizes, int n_in,
                              void* d_out, int out_size, void* d_ws, size_t ws_size,
                              hipStream_t stream) {
  (void)in_sizes; (void)n_in; (void)out_size; (void)ws_size;
  const float* x           = (const float*)d_in[0];
  const float* W_la        = (const float*)d_in[1];
  const float* la_coef     = (const float*)d_in[2];
  const float* kernel_beta = (const float*)d_in[3];
  const float* value_beta  = (const float*)d_in[4];
  const float* W_v         = (const float*)d_in[5];
  const float* v_coef      = (const float*)d_in[6];
  const float* W_proj      = (const float*)d_in[7];
  float* out = (float*)d_out;

  char* ws = (char*)d_ws;
  size_t off = 0;
  auto alloc = [&](size_t bytes) -> char* {
    char* p = ws + off;
    off += (bytes + 255) & ~(size_t)255;
    return p;
  };
  const size_t NTOK = (size_t)B_ * T_;  // 4096 tokens
  _Float16* xh    = (_Float16*)alloc(NTOK * C_ * sizeof(_Float16));
  _Float16* wla_h = (_Float16*)alloc((size_t)C_ * C_ * sizeof(_Float16));
  _Float16* wv_h  = (_Float16*)alloc((size_t)C_ * C_ * sizeof(_Float16));
  _Float16* wpj_h = (_Float16*)alloc((size_t)C_ * C_ * sizeof(_Float16));
  float*    xl    = (float*)alloc(NTOK * C_ * sizeof(float));
  float*    xv    = (float*)alloc(NTOK * C_ * sizeof(float));
  _Float16* kh    = (_Float16*)alloc((size_t)B_ * NH_ * T_ * HS_ * sizeof(_Float16));
  _Float16* vh    = (_Float16*)alloc((size_t)B_ * NH_ * T_ * HS_ * sizeof(_Float16));
  _Float16* yh    = (_Float16*)alloc(NTOK * C_ * sizeof(_Float16));

  const int nx = (int)(NTOK * C_);
  const int nw = C_ * C_;
  cast_f32_to_f16<<<(nx + 255) / 256, 256, 0, stream>>>(x, xh, nx);
  cast_f32_to_f16<<<(nw + 255) / 256, 256, 0, stream>>>(W_la, wla_h, nw);
  cast_f32_to_f16<<<(nw + 255) / 256, 256, 0, stream>>>(W_v, wv_h, nw);
  cast_f32_to_f16<<<(nw + 255) / 256, 256, 0, stream>>>(W_proj, wpj_h, nw);

  dim3 gg(C_ / 128, (unsigned)(NTOK / 128));
  gemm_f16_f32_wmma<<<gg, 256, 0, stream>>>(xh, wla_h, xl, (int)NTOK, C_, C_);
  gemm_f16_f32_wmma<<<gg, 256, 0, stream>>>(xh, wv_h, xv, (int)NTOK, C_, C_);

  ema_norm_k<<<B_ * NH_, 32, 0, stream>>>(xl, la_coef, kernel_beta, kh);

  const int nwaves = B_ * NH_ * T_;
  vmix_norm_k<<<(nwaves * 32 + 255) / 256, 256, 0, stream>>>(xv, v_coef, value_beta, vh);

  dim3 ga(T_ / 64, B_ * NH_);
  flash_attn_wmma<<<ga, 128, 0, stream>>>(kh, vh, yh);

  gemm_f16_f32_wmma<<<gg, 256, 0, stream>>>(yh, wpj_h, out, (int)NTOK, C_, C_);
}
